// SpanClassifier_26448408608915
// MI455X (gfx1250) — matrix operand order repaired and checked by reading, not compiled, not run on hardware
//
#include <hip/hip_runtime.h>
#include <climits>
#include <math.h>

// ---------------------------------------------------------------------------
// BERT span-classifier forward for MI455X (gfx1250), wave32 + WMMA bf16.
// GEMMs: V_WMMA_F32_16X16X32_BF16, 32x64 register-blocked tile per wave
// (8 WMMAs per 6 fragment loads), spill-free register budget (~130 VGPRs).
// ---------------------------------------------------------------------------

typedef __bf16 bf16_t;
typedef __attribute__((ext_vector_type(16))) __bf16 v16bf;
typedef __attribute__((ext_vector_type(8)))  __bf16 v8bf;
typedef __attribute__((ext_vector_type(8)))  float  v8f;

#define Bc   8
#define Sc   512
#define Hc   768
#define NHc  12
#define FFc  3072
#define DHc  64
#define NLc  4
#define NLAB 32
#define MAXSP 8
#define NEGV (-1e30f)

static __device__ inline v8f wmma_bf16(v16bf a, v16bf b, v8f c) {
  return __builtin_amdgcn_wmma_f32_16x16x32_bf16(false, a, false, b, (short)0, c,
                                                 false, false);
}

// A fragment (16x32 bf16): lane holds row M=lane%16; elements 0..7 are
// K = 8*half+e, elements 8..15 are K = 16+8*half+e -> two contiguous 16B loads.
static __device__ inline v16bf load_afrag(const bf16_t* __restrict__ p, int kc, int hf) {
  v8bf x0 = *(const v8bf*)(p + kc + hf * 8);
  v8bf x1 = *(const v8bf*)(p + kc + 16 + hf * 8);
  v16bf a;
#pragma unroll
  for (int i = 0; i < 8; i++) { a[i] = x0[i]; a[i + 8] = x1[i]; }
  return a;
}
// B fragment (32x16 bf16): lane holds col N=lane%16; elements 0..7 are
// K = 8*half+e, elements 8..15 are K = 16+8*half+e -> two contiguous 16B loads.
static __device__ inline v16bf load_bfrag(const bf16_t* __restrict__ p, int kc, int hf) {
  v8bf x0 = *(const v8bf*)(p + kc + hf * 8);
  v8bf x1 = *(const v8bf*)(p + kc + 16 + hf * 8);
  v16bf b;
#pragma unroll
  for (int i = 0; i < 8; i++) { b[i] = x0[i]; b[i + 8] = x1[i]; }
  return b;
}

// ------------------- weight convert + transpose (LDS-tiled) ----------------
// W [K,N] f32 row-major -> WT [N,K] bf16 row-major; 32x32 tile via LDS so
// both global reads and writes are coalesced. K,N multiples of 32.
__global__ void convT_kernel(const float* __restrict__ W, bf16_t* __restrict__ WT,
                             int K, int N) {
  __shared__ float tile[32][33];
  int kt = blockIdx.x, nt = blockIdx.y;
  int tx = threadIdx.x, ty = threadIdx.y;   // (32,8)
#pragma unroll
  for (int r = ty; r < 32; r += 8)
    tile[r][tx] = W[(size_t)(kt * 32 + r) * N + nt * 32 + tx];
  __syncthreads();
#pragma unroll
  for (int r = ty; r < 32; r += 8)
    WT[(size_t)(nt * 32 + r) * K + kt * 32 + tx] = (bf16_t)tile[tx][r];
}

// ------------------------- embedding + layernorm ---------------------------
__global__ void embed_ln_kernel(const int* __restrict__ ids, const int* __restrict__ segs,
                                const float* __restrict__ wemb, const float* __restrict__ pemb,
                                const float* __restrict__ temb, const float* __restrict__ g,
                                const float* __restrict__ bta, float* __restrict__ xf,
                                bf16_t* __restrict__ xb) {
  int tok = blockIdx.x;            // B*S blocks
  int s = tok % Sc;
  int t = threadIdx.x;             // 256
  __shared__ float vec[Hc];
  __shared__ float red[256];
  int id = ids[tok], sg = segs[tok];
  float lsum = 0.f;
  for (int e = t; e < Hc; e += 256) {
    float v = wemb[(size_t)id * Hc + e] + pemb[s * Hc + e] + temb[sg * Hc + e];
    vec[e] = v; lsum += v;
  }
  red[t] = lsum; __syncthreads();
  for (int k = 128; k > 0; k >>= 1) { if (t < k) red[t] += red[t + k]; __syncthreads(); }
  float mean = red[0] / (float)Hc; __syncthreads();
  float lvar = 0.f;
  for (int e = t; e < Hc; e += 256) { float d = vec[e] - mean; lvar += d * d; }
  red[t] = lvar; __syncthreads();
  for (int k = 128; k > 0; k >>= 1) { if (t < k) red[t] += red[t + k]; __syncthreads(); }
  float rstd = rsqrtf(red[0] / (float)Hc + 1e-12f);
  for (int e = t; e < Hc; e += 256) {
    float v = (vec[e] - mean) * rstd * g[e] + bta[e];
    size_t o = (size_t)tok * Hc + e;
    xf[o] = v; xb[o] = (bf16_t)v;
  }
}

// ------------------------- layernorm ---------------------------------------
__global__ void ln_kernel(const float* __restrict__ in, const float* __restrict__ g,
                          const float* __restrict__ bta, float* __restrict__ xf,
                          bf16_t* __restrict__ xb) {
  int tok = blockIdx.x; int t = threadIdx.x;
  __shared__ float vec[Hc];
  __shared__ float red[256];
  float lsum = 0.f;
  for (int e = t; e < Hc; e += 256) { float v = in[(size_t)tok * Hc + e]; vec[e] = v; lsum += v; }
  red[t] = lsum; __syncthreads();
  for (int k = 128; k > 0; k >>= 1) { if (t < k) red[t] += red[t + k]; __syncthreads(); }
  float mean = red[0] / (float)Hc; __syncthreads();
  float lvar = 0.f;
  for (int e = t; e < Hc; e += 256) { float d = vec[e] - mean; lvar += d * d; }
  red[t] = lvar; __syncthreads();
  for (int k = 128; k > 0; k >>= 1) { if (t < k) red[t] += red[t + k]; __syncthreads(); }
  float rstd = rsqrtf(red[0] / (float)Hc + 1e-12f);
  for (int e = t; e < Hc; e += 256) {
    float v = (vec[e] - mean) * rstd * g[e] + bta[e];
    size_t o = (size_t)tok * Hc + e;
    xf[o] = v; xb[o] = (bf16_t)v;
  }
}

// ------------------------- generic WMMA GEMM -------------------------------
// C[M,N] = act(A[M,K]@W + bias [+resid]); A bf16 row-major, WT bf16 [N,K].
// mode: 0 = f32 out, 1 = bf16 out, 2 = bf16 out + exact GELU, 3 = f32 tanh.
// Wave tile: 32x64 (2 M-subtiles x 4 N-subtiles), single fragment buffer
// (spill-free); rows >= M are clamped on load and masked at store; N%64==0.
__global__ void wmma_gemm(const bf16_t* __restrict__ A, const bf16_t* __restrict__ WT,
                          const float* __restrict__ bias, const float* __restrict__ resid,
                          float* __restrict__ outF, bf16_t* __restrict__ outB,
                          int M, int N, int K, int mode) {
  int wave = threadIdx.x >> 5;
  int L = threadIdx.x & 31;
  int lane16 = L & 15, hf = L >> 4;
  int numNT = N >> 6;                  // 64-wide N tiles
  int numMT = (M + 31) >> 5;           // 32-high M tiles
  int tile = blockIdx.x * 8 + wave;
  if (tile >= numMT * numNT) return;
  int mt = tile / numNT, nt = tile % numNT;

  int r0 = mt * 32 + lane16;      if (r0 > M - 1) r0 = M - 1;
  int r1 = mt * 32 + 16 + lane16; if (r1 > M - 1) r1 = M - 1;
  const bf16_t* ap0 = A + (size_t)r0 * K;
  const bf16_t* ap1 = A + (size_t)r1 * K;
  const bf16_t* wp0 = WT + (size_t)(nt * 64 + 0  + lane16) * K;
  const bf16_t* wp1 = WT + (size_t)(nt * 64 + 16 + lane16) * K;
  const bf16_t* wp2 = WT + (size_t)(nt * 64 + 32 + lane16) * K;
  const bf16_t* wp3 = WT + (size_t)(nt * 64 + 48 + lane16) * K;

  v8f c00 = {}, c01 = {}, c02 = {}, c03 = {};
  v8f c10 = {}, c11 = {}, c12 = {}, c13 = {};

  for (int kc = 0; kc < K; kc += 32) {
    __builtin_prefetch(ap0 + kc + 256, 0, 1);       // global_prefetch_b8
    __builtin_prefetch(wp0 + kc + 256, 0, 1);
    v16bf A0 = load_afrag(ap0, kc, hf);
    v16bf A1 = load_afrag(ap1, kc, hf);
    v16bf B0 = load_bfrag(wp0, kc, hf);
    v16bf B1 = load_bfrag(wp1, kc, hf);
    v16bf B2 = load_bfrag(wp2, kc, hf);
    v16bf B3 = load_bfrag(wp3, kc, hf);
    // alternate A fragments so consecutive WMMAs hit different accumulators
    c00 = wmma_bf16(A0, B0, c00); c10 = wmma_bf16(A1, B0, c10);
    c01 = wmma_bf16(A0, B1, c01); c11 = wmma_bf16(A1, B1, c11);
    c02 = wmma_bf16(A0, B2, c02); c12 = wmma_bf16(A1, B2, c12);
    c03 = wmma_bf16(A0, B3, c03); c13 = wmma_bf16(A1, B3, c13);
  }

  auto epilogue = [&](v8f cc, int i, int j) {
    int col = nt * 64 + j * 16 + lane16;
    float bv = bias ? bias[col] : 0.f;
#pragma unroll
    for (int r = 0; r < 8; r++) {
      int row = mt * 32 + i * 16 + r + 8 * hf;     // C: VGPR r -> M = r + 8*half
      if (row >= M) continue;
      size_t o = (size_t)row * N + col;
      float v = cc[r] + bv;
      if (resid) v += resid[o];
      if (mode == 2)      v = 0.5f * v * (1.f + erff(v * 0.70710678118654752f));
      else if (mode == 3) v = tanhf(v);
      if (outF) outF[o] = v;
      if (outB) outB[o] = (bf16_t)v;
    }
  };
  epilogue(c00, 0, 0); epilogue(c01, 0, 1); epilogue(c02, 0, 2); epilogue(c03, 0, 3);
  epilogue(c10, 1, 0); epilogue(c11, 1, 1); epilogue(c12, 1, 2); epilogue(c13, 1, 3);
}

// ---------------- transpose V to [b,h,dh,s] (LDS-tiled) --------------------
__global__ void transpose_v_kernel(const bf16_t* __restrict__ vb, bf16_t* __restrict__ vtb) {
  __shared__ bf16_t tile[32][33];
  int bh = blockIdx.x;               // b*NH + h
  int st = blockIdx.y;               // S/32
  int dt = blockIdx.z;               // DH/32
  int bb = bh / NHc, hh = bh % NHc;
  int tx = threadIdx.x, ty = threadIdx.y;   // (32,8)
#pragma unroll
  for (int r = ty; r < 32; r += 8)
    tile[r][tx] = vb[((size_t)bb * Sc + st * 32 + r) * Hc + hh * DHc + dt * 32 + tx];
  __syncthreads();
#pragma unroll
  for (int r = ty; r < 32; r += 8)
    vtb[((size_t)bh * DHc + dt * 32 + r) * Sc + st * 32 + tx] = tile[tx][r];
}

// ------------------------- fused attention ---------------------------------
// Block = (b, h, 16-row q-tile), 4 waves cooperate:
//  - waves split the 32 key-tiles of QK^T (WMMA -> LDS f32 scores)
//  - waves split the 16 softmax rows (wave-cooperative max/sum)
//  - each wave computes one 16-wide N-tile of P@V (WMMA)
__global__ void attention_kernel(const bf16_t* __restrict__ qb, const bf16_t* __restrict__ kb,
                                 const bf16_t* __restrict__ vtb, const float* __restrict__ masks,
                                 bf16_t* __restrict__ ctxb) {
  int blk = blockIdx.x;
  int qt = blk & 31;                // S/16 = 32
  int h = (blk >> 5) % NHc;
  int b = blk / (32 * NHc);
  int w = threadIdx.x >> 5;         // wave 0..3
  int L = threadIdx.x & 31;
  int lane16 = L & 15, hf = L >> 4;

  __shared__ __align__(32) float  sc[16][Sc];
  __shared__ __align__(32) bf16_t pr[16][Sc];

  // Q fragment for this q-tile (K-dim 64 -> 2 chunks); redundant per wave.
  int qrow = qt * 16 + lane16;
  const bf16_t* qp = qb + ((size_t)b * Sc + qrow) * Hc + h * DHc;
  v16bf aq0 = load_afrag(qp, 0, hf);
  v16bf aq1 = load_afrag(qp, 32, hf);

  // scores = Q @ K^T * scale + mask bias; wave w handles key-tiles w*8..w*8+7
  for (int kt = w * 8; kt < w * 8 + 8; ++kt) {
    int key = kt * 16 + lane16;
    const bf16_t* kp = kb + ((size_t)b * Sc + key) * Hc + h * DHc;
    v16bf b0 = load_bfrag(kp, 0, hf);
    v16bf b1 = load_bfrag(kp, 32, hf);
    v8f acc = {};
    acc = wmma_bf16(aq0, b0, acc);
    acc = wmma_bf16(aq1, b1, acc);
    float biasv = (1.0f - masks[b * Sc + key]) * -10000.0f;  // col N == lane's key
#pragma unroll
    for (int r = 0; r < 8; r++) sc[r + 8 * hf][kt * 16 + lane16] = acc[r] * 0.125f + biasv;
  }
  __syncthreads();
  // softmax: wave w owns rows w*4 .. w*4+3
  for (int row = w * 4; row < w * 4 + 4; ++row) {
    float m = -INFINITY;
    for (int c = L; c < Sc; c += 32) m = fmaxf(m, sc[row][c]);
#pragma unroll
    for (int off = 16; off > 0; off >>= 1) m = fmaxf(m, __shfl_xor(m, off, 32));
    float ssum = 0.f;
    for (int c = L; c < Sc; c += 32) { float e = __expf(sc[row][c] - m); sc[row][c] = e; ssum += e; }
#pragma unroll
    for (int off = 16; off > 0; off >>= 1) ssum += __shfl_xor(ssum, off, 32);
    float inv = 1.0f / ssum;
    for (int c = L; c < Sc; c += 32) pr[row][c] = (bf16_t)(sc[row][c] * inv);
  }
  __syncthreads();
  // ctx = P @ V: wave w computes N-tile w (V pre-transposed: vtb[b,h,dh,s])
  {
    const bf16_t* vtp = vtb + ((size_t)(b * NHc + h) * DHc + w * 16 + lane16) * Sc;
    const bf16_t* prp = &pr[lane16][0];
    v8f acc = {};
    for (int kc = 0; kc < Sc; kc += 32) {
      v16bf a = load_afrag(prp, kc, hf);
      v16bf bv = load_bfrag(vtp, kc, hf);
      acc = wmma_bf16(a, bv, acc);
    }
#pragma unroll
    for (int r = 0; r < 8; r++) {
      int row = qt * 16 + r + 8 * hf;
      ctxb[((size_t)b * Sc + row) * Hc + h * DHc + w * 16 + lane16] = (bf16_t)acc[r];
    }
  }
}

// ------------------------- pooled gather -----------------------------------
__global__ void gather_pooled_kernel(const float* __restrict__ xf, bf16_t* __restrict__ pin) {
  int idx = blockIdx.x * 256 + threadIdx.x;
  if (idx >= Bc * Hc) return;
  int b = idx / Hc, h = idx % Hc;
  pin[idx] = (bf16_t)xf[((size_t)b * Sc) * Hc + h];
}

// ------------------------- start/end GEMV heads ----------------------------
__global__ void startend_kernel(const float* __restrict__ xf, const float* __restrict__ sw,
                                const float* __restrict__ sb, const float* __restrict__ ew,
                                const float* __restrict__ eb, float* __restrict__ outS,
                                float* __restrict__ outE) {
  int tok = blockIdx.x; int t = threadIdx.x;   // 128
  __shared__ float r1[128], r2[128];
  float a = 0.f, c = 0.f;
  for (int h = t; h < Hc; h += 128) {
    float x = xf[(size_t)tok * Hc + h];
    a += x * sw[h]; c += x * ew[h];
  }
  r1[t] = a; r2[t] = c; __syncthreads();
  for (int k = 64; k > 0; k >>= 1) { if (t < k) { r1[t] += r1[t + k]; r2[t] += r2[t + k]; } __syncthreads(); }
  if (t == 0) { outS[tok] = r1[0] + sb[0]; outE[tok] = r2[0] + eb[0]; }
}

// ------------------------- span / n-span heads -----------------------------
__global__ void heads_kernel(const float* __restrict__ pooled, const float* __restrict__ span_w,
                             const float* __restrict__ span_b, const float* __restrict__ nspan_w,
                             const float* __restrict__ nspan_b, float* __restrict__ spans_out,
                             float* __restrict__ nspan_out) {
  int b = blockIdx.x; int t = threadIdx.x;    // 64
  if (t < NLAB) {
    float s = span_b[t];
    for (int h = 0; h < Hc; h++) s += pooled[b * Hc + h] * span_w[h * NLAB + t];
    spans_out[b * NLAB + t] = s;
  } else if (t < NLAB + MAXSP + 1) {
    int c = t - NLAB;
    float s = nspan_b[c];
    for (int h = 0; h < Hc; h++) s += pooled[b * Hc + h] * nspan_w[h * (MAXSP + 1) + c];
    nspan_out[b * (MAXSP + 1) + c] = s;
  }
}

// ------------------------- prediction (greedy NMS, faithful) ---------------
__global__ void predict_kernel(const float* __restrict__ startsL, const float* __restrict__ endsL,
                               const float* __restrict__ spansL, const float* __restrict__ nspanL,
                               const float* __restrict__ xf, const float* __restrict__ span_w,
                               const float* __restrict__ span_b, float* __restrict__ outS,
                               float* __restrict__ outE, float* __restrict__ outL,
                               float* __restrict__ outT) {
  const int b = blockIdx.x, t = threadIdx.x;  // 256 threads
  __shared__ float rv[256]; __shared__ int ri[256];
  __shared__ unsigned char rdead[Sc], cdead[Sc];
  __shared__ int sel_s[MAXSP], sel_e[MAXSP];
  __shared__ int sh_ns, sh_cnt, sh_stop;
  __shared__ float vecs[MAXSP][Hc];
  __shared__ float sl[MAXSP][NLAB];

  const float* st = startsL + b * Sc;
  const float* en = endsL + b * Sc;

  if (t == 0) {
    int bi = 0; float bv = nspanL[b * (MAXSP + 1)];
    for (int i = 1; i < MAXSP + 1; i++) {
      float v = nspanL[b * (MAXSP + 1) + i];
      if (v > bv) { bv = v; bi = i; }
    }
    sh_ns = bi;
  }
  __syncthreads();
  int nsp = sh_ns;

  if (nsp <= 1) {  // single-span branch
    {
      float bv = -INFINITY; int bi = INT_MAX;
      for (int s = t; s < Sc; s += 256) { float v = st[s]; if (v > bv || (v == bv && s < bi)) { bv = v; bi = s; } }
      rv[t] = bv; ri[t] = bi; __syncthreads();
      for (int k = 128; k > 0; k >>= 1) { if (t < k && (rv[t + k] > rv[t] || (rv[t + k] == rv[t] && ri[t + k] < ri[t]))) { rv[t] = rv[t + k]; ri[t] = ri[t + k]; } __syncthreads(); }
      int am = ri[0]; __syncthreads();
      for (int s = t; s < Sc; s += 256) outS[b * Sc + s] = (s == am) ? 1.f : 0.f;
    }
    {
      float bv = -INFINITY; int bi = INT_MAX;
      for (int s = t; s < Sc; s += 256) { float v = en[s]; if (v > bv || (v == bv && s < bi)) { bv = v; bi = s; } }
      rv[t] = bv; ri[t] = bi; __syncthreads();
      for (int k = 128; k > 0; k >>= 1) { if (t < k && (rv[t + k] > rv[t] || (rv[t + k] == rv[t] && ri[t + k] < ri[t]))) { rv[t] = rv[t + k]; ri[t] = ri[t + k]; } __syncthreads(); }
      int am = ri[0]; __syncthreads();
      for (int s = t; s < Sc; s += 256) outE[b * Sc + s] = (s == am) ? 1.f : 0.f;
    }
    if (t == 0) {
      int bi = 0; float bv = spansL[b * NLAB];
      for (int c = 1; c < NLAB; c++) { float v = spansL[b * NLAB + c]; if (v > bv) { bv = v; bi = c; } }
      for (int c = 0; c < NLAB; c++) outL[b * NLAB + c] = (c == bi) ? 1.f : 0.f;
    }
    for (int c = t; c < NLAB; c += 256) outT[b * NLAB + c] = spansL[b * NLAB + c];
    return;
  }

  // multi-span branch: greedy argmax NMS on joint = starts_i * ends_j (i<j)
  for (int i = t; i < Sc; i += 256) { rdead[i] = 0; cdead[i] = 0; }
  if (t < MAXSP) { sel_s[t] = 0; sel_e[t] = 0; }
  if (t == 0) { sh_cnt = 0; sh_stop = 0; }
  __syncthreads();

  for (int iter = 0; iter < Sc + 1; ++iter) {
    float bv = 0.f; int bi = INT_MAX;
    for (int f = t; f < Sc * Sc; f += 256) {
      int i = f >> 9, j = f & (Sc - 1);
      if (j > i && !rdead[i] && !cdead[j]) {
        float v = st[i] * en[j];
        if (v > bv || (v == bv && f < bi)) { bv = v; bi = f; }
      }
    }
    rv[t] = bv; ri[t] = bi; __syncthreads();
    for (int k = 128; k > 0; k >>= 1) { if (t < k && (rv[t + k] > rv[t] || (rv[t + k] == rv[t] && ri[t + k] < ri[t]))) { rv[t] = rv[t + k]; ri[t] = ri[t + k]; } __syncthreads(); }
    if (t == 0) {
      if (sh_cnt >= nsp || !(rv[0] > 0.f)) sh_stop = 1;
      else {
        int f = ri[0]; int s_ = f >> 9, e_ = f & (Sc - 1);
        bool ov = false;
        for (int k = 0; k < sh_cnt; k++) if (s_ <= sel_e[k]) { ov = true; break; }
        if (!ov) { sel_s[sh_cnt] = s_; sel_e[sh_cnt] = e_; sh_cnt++; }
        rdead[s_] = 1; cdead[e_] = 1;
      }
    }
    __syncthreads();
    if (sh_stop) break;
  }
  int cnt = sh_cnt;
  __syncthreads();

  // span pooling: (faithful bug) span k averages tokens of BATCH ROW k
  for (int idx = t; idx < MAXSP * Hc; idx += 256) {
    int k = idx / Hc, h2 = idx % Hc;
    int s0 = sel_s[k], e0 = sel_e[k];
    float sum = 0.f;
    for (int s = s0; s <= e0; ++s) sum += xf[((size_t)k * Sc + s) * Hc + h2];
    vecs[k][h2] = sum / (float)(e0 - s0 + 1);
  }
  __syncthreads();
  {
    int k = t >> 5, c = t & 31;  // 256 = 8 * 32
    float sum = span_b[c];
    for (int h2 = 0; h2 < Hc; ++h2) sum += vecs[k][h2] * span_w[h2 * NLAB + c];
    sl[k][c] = sum;
  }
  __syncthreads();

  for (int s = t; s < Sc; s += 256) { outS[b * Sc + s] = 0.f; outE[b * Sc + s] = 0.f; }
  __syncthreads();
  if (t == 0) {
    float lab[NLAB];
    for (int c = 0; c < NLAB; c++) lab[c] = 0.f;
    for (int k = 0; k < MAXSP; k++) {
      int bi2 = 0; float bv2 = sl[k][0];
      for (int c = 1; c < NLAB; c++) if (sl[k][c] > bv2) { bv2 = sl[k][c]; bi2 = c; }
      float valid = (k < cnt) ? 1.f : 0.f;
      if (valid > lab[bi2]) lab[bi2] = valid;
      int p = sel_s[k];
      if (valid > outS[b * Sc + p]) { outS[b * Sc + p] = valid; outE[b * Sc + p] = valid; }
    }
    for (int c = 0; c < NLAB; c++) outL[b * NLAB + c] = lab[c];
  }
  for (int c = t; c < NLAB; c += 256) {
    float v;
    if (cnt > 0) { v = NEGV; for (int k = 0; k < cnt; k++) v = fmaxf(v, sl[k][c]); }
    else v = spansL[b * NLAB + c];
    outT[b * NLAB + c] = v;
  }
}

// ---------------------------------------------------------------------------
// Host side
// ---------------------------------------------------------------------------
struct LayerPtrs {
  const float *wq, *bq, *wk, *bk, *wv, *bv, *wo, *bo;
  const float *ln1_g, *ln1_b, *w1, *b1, *w2, *b2, *ln2_g, *ln2_b;
};
struct PtrTable {
  const int *input_ids, *segs;
  const float *masks;
  const float *emb_ln_g, *emb_ln_b, *pooler_w, *pooler_b;
  const float *start_w, *start_b, *end_w, *end_b;
  const float *span_w, *span_b, *nspan_w, *nspan_b;
  const float *word_emb, *pos_emb, *type_emb;
  LayerPtrs L[NLc];
};

extern "C" void kernel_launch(void* const* d_in, const int* in_sizes, int n_in,
                              void* d_out, int out_size, void* d_ws, size_t ws_size,
                              hipStream_t stream) {
  (void)n_in; (void)out_size; (void)ws_size;
  PtrTable P;
  // Two plausible flattening conventions for setup_inputs(); detect via the
  // unmistakable word_emb leaf (30522*768 elements).
  const int WEMB = 30522 * Hc;
  bool insertion = (n_in > 3 && in_sizes[3] == WEMB);
  auto F = [&](int i) { return (const float*)d_in[i]; };
  if (insertion) {
    // dict insertion order: input_ids, masks, segs, params{insertion order}
    P.input_ids = (const int*)d_in[0]; P.masks = F(1); P.segs = (const int*)d_in[2];
    P.word_emb = F(3); P.pos_emb = F(4); P.type_emb = F(5);
    P.emb_ln_g = F(6); P.emb_ln_b = F(7);
    P.pooler_w = F(8); P.pooler_b = F(9);
    P.start_w = F(10); P.start_b = F(11); P.end_w = F(12); P.end_b = F(13);
    P.span_w = F(14); P.span_b = F(15); P.nspan_w = F(16); P.nspan_b = F(17);
    for (int l = 0; l < NLc; l++) {
      int o = 18 + 16 * l;
      P.L[l] = { F(o+0), F(o+1), F(o+2), F(o+3), F(o+4), F(o+5), F(o+6), F(o+7),
                 F(o+8), F(o+9), F(o+10), F(o+11), F(o+12), F(o+13), F(o+14), F(o+15) };
    }
  } else {
    // jax pytree order: top-level dict keys sorted; params leaves sorted too.
    // 0 input_ids, 1 masks, [2..80] params(sorted), 81 segs
    P.input_ids = (const int*)d_in[0]; P.masks = F(1); P.segs = (const int*)d_in[81];
    P.emb_ln_b = F(2); P.emb_ln_g = F(3); P.end_b = F(4); P.end_w = F(5);
    for (int l = 0; l < NLc; l++) {
      int o = 6 + 16 * l;   // sorted: b1,b2,bk,bo,bq,bv,ln1_b,ln1_g,ln2_b,ln2_g,w1,w2,wk,wo,wq,wv
      P.L[l].b1 = F(o+0);  P.L[l].b2 = F(o+1);  P.L[l].bk = F(o+2);  P.L[l].bo = F(o+3);
      P.L[l].bq = F(o+4);  P.L[l].bv = F(o+5);  P.L[l].ln1_b = F(o+6); P.L[l].ln1_g = F(o+7);
      P.L[l].ln2_b = F(o+8); P.L[l].ln2_g = F(o+9); P.L[l].w1 = F(o+10); P.L[l].w2 = F(o+11);
      P.L[l].wk = F(o+12); P.L[l].wo = F(o+13); P.L[l].wq = F(o+14); P.L[l].wv = F(o+15);
    }
    P.nspan_b = F(70); P.nspan_w = F(71); P.pooler_b = F(72); P.pooler_w = F(73);
    P.pos_emb = F(74); P.span_b = F(75); P.span_w = F(76);
    P.start_b = F(77); P.start_w = F(78); P.type_emb = F(79); P.word_emb = F(80);
  }

  // -------- workspace layout --------
  char* ws = (char*)d_ws;
  size_t off = 0;
  auto alloc = [&](size_t bytes) { size_t o = off; off = (off + bytes + 255) & ~(size_t)255; return o; };
  const size_t BSH = (size_t)Bc * Sc * Hc;
  float*  xf32   = (float*)(ws + alloc(BSH * 4));
  float*  resb   = (float*)(ws + alloc(BSH * 4));
  bf16_t* xb     = (bf16_t*)(ws + alloc(BSH * 2));
  bf16_t* qb     = (bf16_t*)(ws + alloc(BSH * 2));
  bf16_t* kb     = (bf16_t*)(ws + alloc(BSH * 2));
  bf16_t* vb     = (bf16_t*)(ws + alloc(BSH * 2));
  bf16_t* vtb    = (bf16_t*)(ws + alloc(BSH * 2));
  bf16_t* ctxb   = (bf16_t*)(ws + alloc(BSH * 2));
  bf16_t* h1b    = (bf16_t*)(ws + alloc((size_t)Bc * Sc * FFc * 2));
  bf16_t* wqT    = (bf16_t*)(ws + alloc((size_t)Hc * Hc * 2));
  bf16_t* wkT    = (bf16_t*)(ws + alloc((size_t)Hc * Hc * 2));
  bf16_t* wvT    = (bf16_t*)(ws + alloc((size_t)Hc * Hc * 2));
  bf16_t* woT    = (bf16_t*)(ws + alloc((size_t)Hc * Hc * 2));
  bf16_t* w1T    = (bf16_t*)(ws + alloc((size_t)Hc * FFc * 2));
  bf16_t* w2T    = (bf16_t*)(ws + alloc((size_t)Hc * FFc * 2));
  bf16_t* poolT  = (bf16_t*)(ws + alloc((size_t)Hc * Hc * 2));
  bf16_t* pin    = (bf16_t*)(ws + alloc((size_t)Bc * Hc * 2));
  float*  pooled = (float*)(ws + alloc((size_t)Bc * Hc * 4));
  float*  spansws= (float*)(ws + alloc((size_t)Bc * NLAB * 4));

  float* out = (float*)d_out;
  float* o_start = out;               // [B,S]
  float* o_end   = out + 4096;        // [B,S]
  float* o_spanT = out + 8192;        // [B,32]
  float* o_nspan = out + 8448;        // [B,9]
  float* o_pst   = out + 8520;        // [B,S]
  float* o_pen   = out + 12616;       // [B,S]
  float* o_plab  = out + 16712;       // [B,32]

  auto gemm_blocks = [](int M, int N) {
    int tiles = ((M + 31) / 32) * (N / 64);
    return (tiles + 7) / 8;
  };
  const int TOK = Bc * Sc;
  dim3 tp32(32, 8);

  // embedding + LN
  embed_ln_kernel<<<TOK, 256, 0, stream>>>(P.input_ids, P.segs, P.word_emb, P.pos_emb,
                                           P.type_emb, P.emb_ln_g, P.emb_ln_b, xf32, xb);

  for (int l = 0; l < NLc; l++) {
    const LayerPtrs& Lp = P.L[l];
    convT_kernel<<<dim3(Hc / 32, Hc / 32), tp32, 0, stream>>>(Lp.wq, wqT, Hc, Hc);
    convT_kernel<<<dim3(Hc / 32, Hc / 32), tp32, 0, stream>>>(Lp.wk, wkT, Hc, Hc);
    convT_kernel<<<dim3(Hc / 32, Hc / 32), tp32, 0, stream>>>(Lp.wv, wvT, Hc, Hc);
    convT_kernel<<<dim3(Hc / 32, Hc / 32), tp32, 0, stream>>>(Lp.wo, woT, Hc, Hc);
    convT_kernel<<<dim3(Hc / 32, FFc / 32), tp32, 0, stream>>>(Lp.w1, w1T, Hc, FFc);
    convT_kernel<<<dim3(FFc / 32, Hc / 32), tp32, 0, stream>>>(Lp.w2, w2T, FFc, Hc);

    wmma_gemm<<<gemm_blocks(TOK, Hc), 256, 0, stream>>>(xb, wqT, Lp.bq, nullptr, nullptr, qb, TOK, Hc, Hc, 1);
    wmma_gemm<<<gemm_blocks(TOK, Hc), 256, 0, stream>>>(xb, wkT, Lp.bk, nullptr, nullptr, kb, TOK, Hc, Hc, 1);
    wmma_gemm<<<gemm_blocks(TOK, Hc), 256, 0, stream>>>(xb, wvT, Lp.bv, nullptr, nullptr, vb, TOK, Hc, Hc, 1);

    transpose_v_kernel<<<dim3(Bc * NHc, Sc / 32, DHc / 32), tp32, 0, stream>>>(vb, vtb);
    attention_kernel<<<Bc * NHc * (Sc / 16), 128, 0, stream>>>(qb, kb, vtb, P.masks, ctxb);

    wmma_gemm<<<gemm_blocks(TOK, Hc), 256, 0, stream>>>(ctxb, woT, Lp.bo, xf32, resb, nullptr, TOK, Hc, Hc, 0);
    ln_kernel<<<TOK, 256, 0, stream>>>(resb, Lp.ln1_g, Lp.ln1_b, xf32, xb);

    wmma_gemm<<<gemm_blocks(TOK, FFc), 256, 0, stream>>>(xb, w1T, Lp.b1, nullptr, nullptr, h1b, TOK, FFc, Hc, 2);
    wmma_gemm<<<gemm_blocks(TOK, Hc), 256, 0, stream>>>(h1b, w2T, Lp.b2, xf32, resb, nullptr, TOK, Hc, FFc, 0);
    ln_kernel<<<TOK, 256, 0, stream>>>(resb, Lp.ln2_g, Lp.ln2_b, xf32, xb);
  }

  // pooler: tanh(x[:,0,:] @ pooler_w + b)
  gather_pooled_kernel<<<(Bc * Hc + 255) / 256, 256, 0, stream>>>(xf32, pin);
  convT_kernel<<<dim3(Hc / 32, Hc / 32), tp32, 0, stream>>>(P.pooler_w, poolT, Hc, Hc);
  wmma_gemm<<<gemm_blocks(Bc, Hc), 256, 0, stream>>>(pin, poolT, P.pooler_b, nullptr, pooled, nullptr, Bc, Hc, Hc, 3);

  // heads
  startend_kernel<<<TOK, 128, 0, stream>>>(xf32, P.start_w, P.start_b, P.end_w, P.end_b, o_start, o_end);
  heads_kernel<<<Bc, 64, 0, stream>>>(pooled, P.span_w, P.span_b, P.nspan_w, P.nspan_b, spansws, o_nspan);

  // prediction (greedy NMS etc.)
  predict_kernel<<<Bc, 256, 0, stream>>>(o_start, o_end, spansws, o_nspan, xf32,
                                         P.span_w, P.span_b, o_pst, o_pen, o_plab, o_spanT);
}
